// AttentionNoCache_14534169329961
// MI455X (gfx1250) — matrix operand order repaired and checked by reading, not compiled, not run on hardware
//
#include <hip/hip_runtime.h>
#include <stdint.h>

// CDNA5 / gfx1250: wave32, WMMA 16x16x32 bf16 (f32 accum).
typedef __attribute__((ext_vector_type(16))) __bf16 v16bf;
typedef __attribute__((ext_vector_type(8)))  float  v8f;

#define L_SEQ   2048
#define D_DIM   64
#define NBH     32          // B*H
#define QT      16          // queries per workgroup
#define NWAVE   8
#define TOPK    64
#define SSTRIDE 2064        // padded score-row stride (u32), kills LDS bank conflicts
#define HISTB   256

// monotonic float -> u32 mapping (orders like the floats)
__device__ __forceinline__ unsigned mapf(float s) {
  unsigned b = __float_as_uint(s);
  return (b & 0x80000000u) ? ~b : (b | 0x80000000u);
}
__device__ __forceinline__ float unmapf(unsigned u) {
  unsigned b = (u & 0x80000000u) ? (u ^ 0x80000000u) : ~u;
  return __uint_as_float(b);
}

__global__ __launch_bounds__(NWAVE * 32)
void topk_attn_kernel(const float* __restrict__ Q,
                      const float* __restrict__ K,
                      const float* __restrict__ V,
                      float* __restrict__ Out) {
  extern __shared__ unsigned smem[];
  unsigned* Ssc  = smem;                         // QT * SSTRIDE  (mapped scores)
  unsigned* Hist = Ssc + QT * SSTRIDE;           // QT * 256
  unsigned* LidX = Hist + QT * HISTB;            // QT * TOPK (selected key idx)
  float*    LwW  = (float*)(LidX + QT * TOPK);   // QT * TOPK (softmax weights)
  unsigned* Cnt  = (unsigned*)(LwW + QT * TOPK); // QT * 2   (nsel, tiecnt)

  const int tid  = threadIdx.x;
  const int lane = tid & 31;
  const int wave = tid >> 5;
  const int bh   = blockIdx.x >> 7;              // 128 q-tiles per head
  const int qt   = blockIdx.x & 127;
  const int q0   = qt * QT;

  const size_t headOff = (size_t)bh * L_SEQ * D_DIM;
  const float* Qh = Q + headOff;
  const float* Kh = K + headOff;
  const float* Vh = V + headOff;

  // ================= Phase 1: S = Q K^T via bf16 WMMA =================
  // A fragment (16x32 bf16): lane m=lane&15, khalf=lane>>4;
  // elem e -> k = khalf*16 + e
  const int am = lane & 15;
  const int kh = lane >> 4;
  v16bf a0, a1;
  {
    const float* qrow = Qh + (size_t)(q0 + am) * D_DIM;
#pragma unroll
    for (int e = 0; e < 16; ++e) {
      const int k = (kh << 4) + e;
      a0[e] = (__bf16)qrow[k];          // dims [0,32)
      a1[e] = (__bf16)qrow[k + 32];     // dims [32,64)
    }
  }

  // B fragment (32x16 bf16): lane n=lane&15, kgroup=lane>>4; elem e -> k = kgroup*16 + e
  // B[k][n] = K[key n][dim k] -> 16 contiguous floats per lane (no transpose needed).
  const int bn = lane & 15;
  const int kg = lane >> 4;
#pragma unroll 2
  for (int t = 0; t < 16; ++t) {
    const int kbase = (wave * 16 + t) * 16;
    const float* krow = Kh + (size_t)(kbase + bn) * D_DIM + (kg << 4);
    if (t + 1 < 16)
      __builtin_prefetch(krow + 16 * D_DIM, 0, 1);   // global_prefetch_b8: next K tile

    v16bf b0, b1;
#pragma unroll
    for (int e = 0; e < 16; ++e) {
      b0[e] = (__bf16)krow[e];          // dims [0,32)
      b1[e] = (__bf16)krow[e + 32];     // dims [32,64)
    }
    v8f c = {};
    c = __builtin_amdgcn_wmma_f32_16x16x32_bf16(false, a0, false, b0,
                                                (short)0, c, false, false);
    c = __builtin_amdgcn_wmma_f32_16x16x32_bf16(false, a1, false, b1,
                                                (short)0, c, false, false);
    // C/D layout: lane n=lane&15; VGPR r holds m = r + 8*(lane>>4)
    const int col = kbase + bn;
    const int mb  = (lane >> 4) << 3;
#pragma unroll
    for (int r = 0; r < 8; ++r)
      Ssc[(mb + r) * SSTRIDE + col] = mapf(c[r]);
  }

  __syncthreads();

  // ========= Phase 2: exact top-64 (4-pass radix select) + softmax + AV =========
  for (int qq = 0; qq < 2; ++qq) {
    const int q = wave * 2 + qq;                 // one wave owns this query's LDS state
    unsigned* Srow  = Ssc  + q * SSTRIDE;
    unsigned* hist  = Hist + q * HISTB;
    unsigned* listI = LidX + q * TOPK;
    float*    listW = LwW  + q * TOPK;
    unsigned* cnts  = Cnt  + q * 2;

    unsigned prefix = 0, krem = TOPK, umax = 0u;

#pragma unroll 1
    for (int pass = 0; pass < 4; ++pass) {
      const int shift = 24 - 8 * pass;
#pragma unroll
      for (int j = 0; j < 8; ++j) hist[lane * 8 + j] = 0u;   // same-wave LDS is in-order
#pragma unroll 4
      for (int i = 0; i < 64; ++i) {
        const unsigned u = Srow[lane + (i << 5)];
        if (pass == 0) {
          umax = u > umax ? u : umax;
          atomicAdd(&hist[u >> 24], 1u);
        } else if ((u >> (shift + 8)) == prefix) {
          atomicAdd(&hist[(u >> shift) & 255u], 1u);
        }
      }
      // lane l owns bins [8l, 8l+8); suffix-sum counts from high bins down
      unsigned h[8], cnt = 0;
#pragma unroll
      for (int j = 0; j < 8; ++j) { h[j] = hist[lane * 8 + j]; cnt += h[j]; }
      unsigned suf = cnt;
#pragma unroll
      for (int off = 1; off < 32; off <<= 1) {
        const unsigned t2 = __shfl_down(suf, off, 32);
        if (lane + off < 32) suf += t2;
      }
      const unsigned above = suf - cnt;          // count with byte > lane's top bin
      unsigned packed = 0;
      if (above < krem && suf >= krem) {         // exactly one boundary lane
        unsigned run = above;
#pragma unroll
        for (int j = 7; j >= 0; --j) {
          if (run + h[j] >= krem) {
            packed = ((unsigned)(lane * 8 + j) << 16) | (krem - run);
            break;
          }
          run += h[j];
        }
      }
#pragma unroll
      for (int m = 16; m >= 1; m >>= 1) packed |= __shfl_xor(packed, m, 32);
      prefix = (prefix << 8) | (packed >> 16);
      krem   = packed & 0xFFFFu;
    }
    const unsigned T = prefix;                   // exact 64th-largest mapped score

#pragma unroll
    for (int m = 16; m >= 1; m >>= 1) {          // row max for stable softmax
      const unsigned t2 = __shfl_xor(umax, m, 32);
      umax = t2 > umax ? t2 : umax;
    }
    const float smax = unmapf(umax);

    if (lane == 0) { cnts[0] = 0u; cnts[1] = 0u; }
    float dsum = 0.0f;
#pragma unroll 4
    for (int i = 0; i < 64; ++i) {
      const int idx = lane + (i << 5);
      const unsigned u = Srow[idx];
      bool sel = u > T;
      if (!sel && u == T)
        sel = atomicAdd(&cnts[1], 1u) < krem;    // exact tie resolution -> 64 total
      if (sel) {
        const float w = __expf(unmapf(u) - smax);
        const unsigned slot = atomicAdd(&cnts[0], 1u);
        if (slot < TOPK) { listI[slot] = (unsigned)idx; listW[slot] = w; dsum += w; }
      }
    }
#pragma unroll
    for (int m = 16; m >= 1; m >>= 1) dsum += __shfl_xor(dsum, m, 32);

    unsigned nsel = cnts[0];
    nsel = nsel > TOPK ? TOPK : nsel;
    float acc0 = 0.0f, acc1 = 0.0f;              // lane covers out dims l and l+32
#pragma unroll 1
    for (unsigned e = 0; e < nsel; ++e) {
      const unsigned idx = listI[e];             // LDS broadcast
      const float    w   = listW[e];
      const float* vr = Vh + (size_t)idx * D_DIM;
      acc0 = fmaf(w, vr[lane], acc0);
      acc1 = fmaf(w, vr[lane + 32], acc1);
    }
    const float inv = 1.0f / dsum;
    float* orow = Out + headOff + (size_t)(q0 + q) * D_DIM;
    orow[lane]      = acc0 * inv;
    orow[lane + 32] = acc1 * inv;
  }
}

extern "C" void kernel_launch(void* const* d_in, const int* in_sizes, int n_in,
                              void* d_out, int out_size, void* d_ws, size_t ws_size,
                              hipStream_t stream) {
  (void)in_sizes; (void)n_in; (void)out_size; (void)d_ws; (void)ws_size;
  const float* Q = (const float*)d_in[0];
  const float* K = (const float*)d_in[1];
  const float* V = (const float*)d_in[2];
  float* Out = (float*)d_out;                    // d_in[3] = topk (compile-time 64)

  const dim3 grid(NBH * (L_SEQ / QT));           // 4096 workgroups
  const dim3 block(NWAVE * 32);                  // 8 wave32s
  const size_t shmem =
      (size_t)(QT * SSTRIDE + QT * HISTB + QT * TOPK) * sizeof(unsigned) +
      (size_t)(QT * TOPK) * sizeof(float) + (size_t)(QT * 2) * sizeof(unsigned);
  topk_attn_kernel<<<grid, block, shmem, stream>>>(Q, K, V, Out);
}
